// CausalSelfAttention_26680336842833
// MI455X (gfx1250) — matrix operand order repaired and checked by reading, not compiled, not run on hardware
//
#include <hip/hip_runtime.h>

typedef __attribute__((ext_vector_type(16))) _Float16 v16h;
typedef __attribute__((ext_vector_type(8)))  _Float16 v8h;
typedef __attribute__((ext_vector_type(8)))  float    v8f;

#define N_EMBD 1024
#define N_HEAD 16
#define HD     64
#define B_     2
#define T_     2048
#define M_TOT  (B_ * T_)          // 4096
#define LOG2E  1.44269504088896f
#define QSCALE (0.125f * LOG2E)   // 1/sqrt(64) * log2(e), folded into Q

#define TK      64                 // k-depth per LDS stage (GEMM)
#define NSTAGE  (N_EMBD / TK)      // 16 stages
#define LDP     72                 // padded row stride (halves): 36-dword stride, conflict-free

#define WMMA(a, b, c) \
  __builtin_amdgcn_wmma_f32_16x16x32_f16(false, (a), false, (b), (short)0, (c), false, false)

// ---------------------------------------------------------------------------
// CDNA5 async global->LDS copy (ASYNCcnt-tracked).  LDS byte address passed as
// a 32-bit VGPR (truncated generic pointer == wave-relative LDS offset).
// ---------------------------------------------------------------------------
__device__ __forceinline__ void async_b128(const _Float16* g, const _Float16* lds) {
  asm volatile("global_load_async_to_lds_b128 %0, %1, off"
               :: "v"((unsigned)(size_t)lds), "v"(g) : "memory");
}

template <int N>
__device__ __forceinline__ void wait_async() {
#if __has_builtin(__builtin_amdgcn_s_wait_asynccnt)
  __builtin_amdgcn_s_wait_asynccnt(N);
#else
  asm volatile("s_wait_asynccnt %0" :: "i"(N) : "memory");
#endif
}

// ---------------------------------------------------------------------------
// Fragment loaders (wave32, v_wmma_f32_16x16x32_f16 layouts per ISA 7.12.2)
// A-frag: lane L: m = L%16; halves 0..7 = A[m][kb+(L/16)*8+0..7],
//                           halves 8..15 = A[m][kb+16+(L/16)*8+0..7]
// B-frag (B = Wm^T, Wm row-major [N][K]): lane L: n = L%16;
//         halves 0..7 at k = kb+(L/16)*8+0..7, halves 8..15 at k = kb+16+(L/16)*8+0..7
// C/D frag: lane L holds n = L%16; VGPR r holds m = r + 8*(L/16)
// ---------------------------------------------------------------------------
__device__ __forceinline__ v16h load_a_frag(const _Float16* __restrict__ A, int ldk,
                                            int row0, int kb, int lane) {
  int m  = row0 + (lane & 15);
  int ks = (lane >> 4) * 8;
  const _Float16* p = A + (size_t)m * ldk + kb + ks;
  v8h lo = *(const v8h*)(p);
  v8h hi = *(const v8h*)(p + 16);
  return __builtin_shufflevector(lo, hi, 0,1,2,3,4,5,6,7,8,9,10,11,12,13,14,15);
}

__device__ __forceinline__ v16h load_b_frag(const _Float16* __restrict__ Wm, int ldk,
                                            int col0, int kb, int lane) {
  int n  = col0 + (lane & 15);
  int k0 = kb + (lane >> 4) * 8;
  const _Float16* p = Wm + (size_t)n * ldk + k0;
  v8h lo = *(const v8h*)(p);
  v8h hi = *(const v8h*)(p + 16);
  return __builtin_shufflevector(lo, hi, 0,1,2,3,4,5,6,7,8,9,10,11,12,13,14,15);
}

// Same layouts read from an LDS tile (row stride LDP halves)
__device__ __forceinline__ v16h lds_a_frag(const _Float16* As, int row0, int kb, int lane) {
  int m  = row0 + (lane & 15);
  int ks = (lane >> 4) * 8;
  const _Float16* p = As + m * LDP + kb + ks;
  v8h lo = *(const v8h*)(p);
  v8h hi = *(const v8h*)(p + 16);
  return __builtin_shufflevector(lo, hi, 0,1,2,3,4,5,6,7,8,9,10,11,12,13,14,15);
}

__device__ __forceinline__ v16h lds_b_frag(const _Float16* Bs, int col0, int kb, int lane) {
  int n  = col0 + (lane & 15);
  int k0 = kb + (lane >> 4) * 8;
  const _Float16* p = Bs + n * LDP + k0;
  v8h lo = *(const v8h*)(p);
  v8h hi = *(const v8h*)(p + 16);
  return __builtin_shufflevector(lo, hi, 0,1,2,3,4,5,6,7,8,9,10,11,12,13,14,15);
}

// ---------------------------------------------------------------------------
// Cooperative stage copy: 128x64 halves each for A and B, 8 async b128/thread.
// ---------------------------------------------------------------------------
__device__ __forceinline__ void issue_stage(const _Float16* __restrict__ Ag,
                                            const _Float16* __restrict__ Wg,
                                            _Float16* As, _Float16* Bs,
                                            int mtile, int ntile, int kb, int tid) {
#pragma unroll
  for (int r = 0; r < 4; r++) {
    int c   = tid + r * 256;   // 0..1023 chunks of 8 halves
    int row = c >> 3;
    int ci  = (c & 7) * 8;
    async_b128(Ag + (size_t)(mtile + row) * N_EMBD + kb + ci, As + row * LDP + ci);
    async_b128(Wg + (size_t)(ntile + row) * N_EMBD + kb + ci, Bs + row * LDP + ci);
  }
}

__device__ __forceinline__ void compute_stage(const _Float16* As, const _Float16* Bs,
                                              v8f acc[2][4], int am, int bn, int lane) {
#pragma unroll
  for (int kk = 0; kk < TK; kk += 32) {
    v16h a0 = lds_a_frag(As, am,      kk, lane);
    v16h a1 = lds_a_frag(As, am + 16, kk, lane);
    v16h b0 = lds_b_frag(Bs, bn,      kk, lane);
    v16h b1 = lds_b_frag(Bs, bn + 16, kk, lane);
    v16h b2 = lds_b_frag(Bs, bn + 32, kk, lane);
    v16h b3 = lds_b_frag(Bs, bn + 48, kk, lane);
    acc[0][0] = WMMA(a0, b0, acc[0][0]);
    acc[0][1] = WMMA(a0, b1, acc[0][1]);
    acc[0][2] = WMMA(a0, b2, acc[0][2]);
    acc[0][3] = WMMA(a0, b3, acc[0][3]);
    acc[1][0] = WMMA(a1, b0, acc[1][0]);
    acc[1][1] = WMMA(a1, b1, acc[1][1]);
    acc[1][2] = WMMA(a1, b2, acc[1][2]);
    acc[1][3] = WMMA(a1, b3, acc[1][3]);
  }
}

// ---------------------------------------------------------------------------
// fp32 -> f16 conversion
// ---------------------------------------------------------------------------
__global__ void f32_to_f16(const float* __restrict__ src, _Float16* __restrict__ dst, int n) {
  int i = blockIdx.x * blockDim.x + threadIdx.x;
  if (i < n) dst[i] = (_Float16)src[i];
}

// ---------------------------------------------------------------------------
// QKV projection: out = x @ W^T + b, LDS double-buffered via async loads.
//   blockIdx.z: 0 -> Q [b,h,t,d] (pre-scaled by QSCALE), 1 -> K [b,h,t,d],
//               2 -> V^T [b,h,d,t]
// WG = 256 threads = 8 waves; wave tile 32x64; WG tile 128x128.
// ---------------------------------------------------------------------------
__global__ __launch_bounds__(256) void gemm_qkv(
    const _Float16* __restrict__ xh,
    const _Float16* __restrict__ wqh, const _Float16* __restrict__ wkh,
    const _Float16* __restrict__ wvh,
    const float* __restrict__ bq, const float* __restrict__ bk, const float* __restrict__ bv,
    _Float16* __restrict__ Qh, _Float16* __restrict__ Kh, _Float16* __restrict__ VT) {
  __shared__ __align__(16) _Float16 As[2][128 * LDP];
  __shared__ __align__(16) _Float16 Bs[2][128 * LDP];

  int tid  = threadIdx.x;
  int lane = tid & 31;
  int wave = tid >> 5;
  int proj = blockIdx.z;
  const _Float16* W   = (proj == 0) ? wqh : (proj == 1) ? wkh : wvh;
  const float*   bias = (proj == 0) ? bq  : (proj == 1) ? bk  : bv;

  int mtile = blockIdx.x * 128;
  int ntile = blockIdx.y * 128;
  int am = (wave & 3) * 32;
  int bn = (wave >> 2) * 64;

  v8f acc[2][4];
#pragma unroll
  for (int i = 0; i < 2; i++)
#pragma unroll
    for (int j = 0; j < 4; j++) acc[i][j] = {};

  issue_stage(xh, W, As[0], Bs[0], mtile, ntile, 0,  tid);
  issue_stage(xh, W, As[1], Bs[1], mtile, ntile, TK, tid);

  for (int i = 0; i < NSTAGE; i++) {
    int buf = i & 1;
    if (i < NSTAGE - 1) wait_async<8>(); else wait_async<0>();
    __syncthreads();
    compute_stage(As[buf], Bs[buf], acc, am, bn, lane);
    __syncthreads();
    if (i + 2 < NSTAGE)
      issue_stage(xh, W, As[buf], Bs[buf], mtile, ntile, (i + 2) * TK, tid);
  }

  int hi = lane >> 4;
  int nc = lane & 15;
#pragma unroll
  for (int mi = 0; mi < 2; mi++)
#pragma unroll
    for (int ni = 0; ni < 4; ni++)
#pragma unroll
      for (int r = 0; r < 8; r++) {
        int m = mtile + am + mi * 16 + r + 8 * hi;
        int n = ntile + bn + ni * 16 + nc;
        float v = acc[mi][ni][r] + bias[n];
        int bb = m >> 11, t = m & (T_ - 1);
        int h  = n >> 6,  d = n & (HD - 1);
        if (proj == 2) {
          VT[((size_t)(bb * N_HEAD + h) * HD + d) * T_ + t] = (_Float16)v;
        } else if (proj == 0) {
          // fold softmax scale & log2(e) into Q so logits are exp2-ready
          Qh[((size_t)(bb * N_HEAD + h) * T_ + t) * HD + d] = (_Float16)(v * QSCALE);
        } else {
          Kh[((size_t)(bb * N_HEAD + h) * T_ + t) * HD + d] = (_Float16)v;
        }
      }
}

// ---------------------------------------------------------------------------
// Flash attention: one wave owns 16 query rows; Bc = 64 key block
// (softmax bookkeeping amortized over twice the columns per block).
// grid = (T/128, H, B), block = 256 (8 waves -> 128 query rows per WG).
// ---------------------------------------------------------------------------
__global__ __launch_bounds__(256) void attn(
    const _Float16* __restrict__ Qh, const _Float16* __restrict__ Kh,
    const _Float16* __restrict__ VT, _Float16* __restrict__ Yh) {
  __shared__ __align__(16) _Float16 plds[8][16][72];  // per-wave 16x64 P tile, padded

  int lane = threadIdx.x & 31;
  int wave = threadIdx.x >> 5;
  int h = blockIdx.y, b = blockIdx.z;
  int qb = blockIdx.x * 128 + wave * 16;

  const _Float16* Qp = Qh + (size_t)(b * N_HEAD + h) * T_ * HD;
  const _Float16* Kp = Kh + (size_t)(b * N_HEAD + h) * T_ * HD;
  const _Float16* Vp = VT + (size_t)(b * N_HEAD + h) * HD * T_;

  v16h qa0 = load_a_frag(Qp, HD, qb, 0, lane);   // Q pre-scaled by QSCALE
  v16h qa1 = load_a_frag(Qp, HD, qb, 32, lane);

  v8f o[4];
#pragma unroll
  for (int nt = 0; nt < 4; nt++) o[nt] = {};
  float mrow[8], lrow[8];
#pragma unroll
  for (int r = 0; r < 8; r++) { mrow[r] = -1e30f; lrow[r] = 0.f; }

  int hi = lane >> 4;
  int nc = lane & 15;
  int nkb = (qb + 79) >> 6;  // ceil((qb+16)/64): causal block count

  for (int kb = 0; kb < nkb; kb++) {
    int k0 = kb * 64;

    // S = Q @ K^T  (four 16x16 S tiles across 64 key columns; K-dim = 64)
    v8f s[4];
#pragma unroll
    for (int nt = 0; nt < 4; nt++) {
      s[nt] = {};
      v16h kf0 = load_b_frag(Kp, HD, k0 + nt * 16, 0,  lane);
      v16h kf1 = load_b_frag(Kp, HD, k0 + nt * 16, 32, lane);
      s[nt] = WMMA(qa0, kf0, s[nt]);
      s[nt] = WMMA(qa1, kf1, s[nt]);
    }

    // causal mask (logits already in exp2 domain thanks to pre-scaled Q)
    float z[4][8];
#pragma unroll
    for (int nt = 0; nt < 4; nt++)
#pragma unroll
      for (int r = 0; r < 8; r++) {
        int m = qb + r + 8 * hi;
        z[nt][r] = (k0 + nt * 16 + nc <= m) ? s[nt][r] : -1e30f;
      }

    // online softmax, per row (rows live in 16-lane halves -> xor 1,2,4,8)
#pragma unroll
    for (int r = 0; r < 8; r++) {
      float rmax = fmaxf(fmaxf(z[0][r], z[1][r]), fmaxf(z[2][r], z[3][r]));
      rmax = fmaxf(rmax, __shfl_xor(rmax, 1, 32));
      rmax = fmaxf(rmax, __shfl_xor(rmax, 2, 32));
      rmax = fmaxf(rmax, __shfl_xor(rmax, 4, 32));
      rmax = fmaxf(rmax, __shfl_xor(rmax, 8, 32));
      float mnew  = fmaxf(mrow[r], rmax);
      float alpha = exp2f(mrow[r] - mnew);
      float p0 = exp2f(z[0][r] - mnew);
      float p1 = exp2f(z[1][r] - mnew);
      float p2 = exp2f(z[2][r] - mnew);
      float p3 = exp2f(z[3][r] - mnew);
      float rs = (p0 + p1) + (p2 + p3);
      rs += __shfl_xor(rs, 1, 32);
      rs += __shfl_xor(rs, 2, 32);
      rs += __shfl_xor(rs, 4, 32);
      rs += __shfl_xor(rs, 8, 32);
      lrow[r] = lrow[r] * alpha + rs;
      mrow[r] = mnew;
#pragma unroll
      for (int nt = 0; nt < 4; nt++) o[nt][r] *= alpha;
      plds[wave][r + 8 * hi][nc]      = (_Float16)p0;
      plds[wave][r + 8 * hi][16 + nc] = (_Float16)p1;
      plds[wave][r + 8 * hi][32 + nc] = (_Float16)p2;
      plds[wave][r + 8 * hi][48 + nc] = (_Float16)p3;
    }
    __builtin_amdgcn_wave_barrier();  // LDS in-order per wave; fence compiler

    // P (C-layout in LDS) -> two A-frags (k = 0..31, 32..63)
    const _Float16* prow = &plds[wave][lane & 15][0];
    int ks = (lane >> 4) * 8;
    v8h l0 = *(const v8h*)(prow + ks);
    v8h h0 = *(const v8h*)(prow + 16 + ks);
    v8h l1 = *(const v8h*)(prow + 32 + ks);
    v8h h1 = *(const v8h*)(prow + 48 + ks);
    v16h pa0 = __builtin_shufflevector(l0, h0, 0,1,2,3,4,5,6,7,8,9,10,11,12,13,14,15);
    v16h pa1 = __builtin_shufflevector(l1, h1, 0,1,2,3,4,5,6,7,8,9,10,11,12,13,14,15);

    // O += P @ V  (V^T rows are d, contiguous in t)
#pragma unroll
    for (int nt = 0; nt < 4; nt++) {
      v16h vb0 = load_b_frag(Vp, T_, nt * 16, k0,      lane);
      v16h vb1 = load_b_frag(Vp, T_, nt * 16, k0 + 32, lane);
      o[nt] = WMMA(pa0, vb0, o[nt]);
      o[nt] = WMMA(pa1, vb1, o[nt]);
    }
    __builtin_amdgcn_wave_barrier();
  }

  // normalize and scatter to y [b, t, h*64 + d] (f16 for final GEMM)
#pragma unroll
  for (int nt = 0; nt < 4; nt++)
#pragma unroll
    for (int r = 0; r < 8; r++) {
      int t = qb + r + 8 * hi;
      int d = nt * 16 + nc;
      float v = o[nt][r] / lrow[r];
      Yh[(size_t)(b * T_ + t) * N_EMBD + h * HD + d] = (_Float16)v;
    }
}

// ---------------------------------------------------------------------------
// Output projection: out = y @ Wp^T + bp (fp32 result), same LDS pipeline.
// ---------------------------------------------------------------------------
__global__ __launch_bounds__(256) void gemm_out(
    const _Float16* __restrict__ Yh, const _Float16* __restrict__ wph,
    const float* __restrict__ bp, float* __restrict__ out) {
  __shared__ __align__(16) _Float16 As[2][128 * LDP];
  __shared__ __align__(16) _Float16 Bs[2][128 * LDP];

  int tid  = threadIdx.x;
  int lane = tid & 31;
  int wave = tid >> 5;
  int mtile = blockIdx.x * 128;
  int ntile = blockIdx.y * 128;
  int am = (wave & 3) * 32;
  int bn = (wave >> 2) * 64;

  v8f acc[2][4];
#pragma unroll
  for (int i = 0; i < 2; i++)
#pragma unroll
    for (int j = 0; j < 4; j++) acc[i][j] = {};

  issue_stage(Yh, wph, As[0], Bs[0], mtile, ntile, 0,  tid);
  issue_stage(Yh, wph, As[1], Bs[1], mtile, ntile, TK, tid);

  for (int i = 0; i < NSTAGE; i++) {
    int buf = i & 1;
    if (i < NSTAGE - 1) wait_async<8>(); else wait_async<0>();
    __syncthreads();
    compute_stage(As[buf], Bs[buf], acc, am, bn, lane);
    __syncthreads();
    if (i + 2 < NSTAGE)
      issue_stage(Yh, wph, As[buf], Bs[buf], mtile, ntile, (i + 2) * TK, tid);
  }

  int hi = lane >> 4;
  int nc = lane & 15;
#pragma unroll
  for (int mi = 0; mi < 2; mi++)
#pragma unroll
    for (int ni = 0; ni < 4; ni++)
#pragma unroll
      for (int r = 0; r < 8; r++) {
        int m = mtile + am + mi * 16 + r + 8 * hi;
        int n = ntile + bn + ni * 16 + nc;
        out[(size_t)m * N_EMBD + n] = acc[mi][ni][r] + bp[n];
      }
}

// ---------------------------------------------------------------------------
extern "C" void kernel_launch(void* const* d_in, const int* in_sizes, int n_in,
                              void* d_out, int out_size, void* d_ws, size_t ws_size,
                              hipStream_t stream) {
  const float* x  = (const float*)d_in[0];
  const float* Wq = (const float*)d_in[1];
  const float* bq = (const float*)d_in[2];
  const float* Wk = (const float*)d_in[3];
  const float* bk = (const float*)d_in[4];
  const float* Wv = (const float*)d_in[5];
  const float* bv = (const float*)d_in[6];
  const float* Wp = (const float*)d_in[7];
  const float* bp = (const float*)d_in[8];
  float* out = (float*)d_out;

  char* ws = (char*)d_ws;
  const size_t MB = 1u << 20;
  _Float16* xh  = (_Float16*)(ws + 0 * MB);   // 8 MB  (4096x1024)
  _Float16* wqh = (_Float16*)(ws + 8 * MB);   // 2 MB each
  _Float16* wkh = (_Float16*)(ws + 10 * MB);
  _Float16* wvh = (_Float16*)(ws + 12 * MB);
  _Float16* wph = (_Float16*)(ws + 14 * MB);
  _Float16* Qh  = (_Float16*)(ws + 16 * MB);  // 8 MB [b,h,t,d], pre-scaled
  _Float16* Kh  = (_Float16*)(ws + 24 * MB);  // 8 MB [b,h,t,d]
  _Float16* VT  = (_Float16*)(ws + 32 * MB);  // 8 MB [b,h,d,t]
  _Float16* Yh  = (_Float16*)(ws + 40 * MB);  // 8 MB [b*t, c]

  int nx = M_TOT * N_EMBD;        // 4 Mi
  int nw = N_EMBD * N_EMBD;       // 1 Mi
  f32_to_f16<<<(nx + 255) / 256, 256, 0, stream>>>(x,  xh,  nx);
  f32_to_f16<<<(nw + 255) / 256, 256, 0, stream>>>(Wq, wqh, nw);
  f32_to_f16<<<(nw + 255) / 256, 256, 0, stream>>>(Wk, wkh, nw);
  f32_to_f16<<<(nw + 255) / 256, 256, 0, stream>>>(Wv, wvh, nw);
  f32_to_f16<<<(nw + 255) / 256, 256, 0, stream>>>(Wp, wph, nw);

  gemm_qkv<<<dim3(M_TOT / 128, N_EMBD / 128, 3), 256, 0, stream>>>(
      xh, wqh, wkh, wvh, bq, bk, bv, Qh, Kh, VT);

  attn<<<dim3(T_ / 128, N_HEAD, B_), 256, 0, stream>>>(Qh, Kh, VT, Yh);

  gemm_out<<<dim3(M_TOT / 128, N_EMBD / 128, 1), 256, 0, stream>>>(Yh, wph, bp, out);
}
